// LFQ_45148696216374
// MI455X (gfx1250) — matrix operand order repaired and checked by reading, not compiled, not run on hardware
//
#include <hip/hip_runtime.h>
#include <hip/hip_fp16.h>

#define DIM       128
#define NCODES    16384
#define NTOKENS   16384
#define MTILE     128      // tokens per workgroup
#define NCHUNK    128      // codebook rows staged in LDS per step
#define NSPLIT    8        // grid.y split over the codebook
#define THREADS   256      // 8 wave32s
#define COLSPAN   (NCODES / NSPLIT)
#define NCHUNKS   (COLSPAN / NCHUNK)

typedef __attribute__((ext_vector_type(16))) _Float16 v16h;
typedef __attribute__((ext_vector_type(8)))  float    v8f;

union Frag {
  v16h v;
  unsigned u[8];
  uint4 q[2];
};

// pack two f32 -> two f16 (RNE) in one dword
__device__ __forceinline__ unsigned pack_f16(float lo, float hi) {
  __half2 h = __floats2half2_rn(lo, hi);
  return *(unsigned*)&h;
}

// ---- CDNA5 async global->LDS copy (ASYNCcnt-tracked), per ISA 15.18.3 ----
__device__ __forceinline__ void async_copy_b128(unsigned lds_off, const void* gptr) {
  asm volatile("global_load_async_to_lds_b128 %0, %1, off"
               :: "v"(lds_off), "v"(gptr) : "memory");
}
__device__ __forceinline__ void wait_asynccnt0() {
  asm volatile("s_wait_asynccnt 0" ::: "memory");
}

// =====================================================================
// Main kernel: double-buffered async-LDS staging of a pre-converted f16
// codebook, persistent A fragments, fused WMMA + argmax.
// =====================================================================
__global__ __launch_bounds__(THREADS)
void lfq_wmma_argmax_async(const float* __restrict__ x,
                           const __half* __restrict__ cbh,
                           unsigned long long* __restrict__ ws) {
  // [0,32K) = B buffer 0; [32K,64K) = x-tile staging, recycled as B buffer 1
  __shared__ unsigned char ldsMem[65536];

  const int tid  = threadIdx.x;
  const int wave = tid >> 5;
  const int lane = tid & 31;
  const int h    = lane >> 4;   // half-wave
  const int m    = lane & 15;   // row (A) / column (B,C) slot

  const int rowBase  = blockIdx.x * MTILE + wave * 16;
  const int colStart = blockIdx.y * COLSPAN;

  const unsigned ldsBase = (unsigned)(uintptr_t)&ldsMem[0];

  // ---- kick off async copy of first codebook chunk into buf0 ----
  {
    const char* g = (const char*)(cbh + (size_t)colStart * DIM);
#pragma unroll
    for (int r = 0; r < 8; ++r)
      async_copy_b128(ldsBase + tid * 16 + r * 4096, g + tid * 16 + r * 4096);
  }

  // ---- stage this wave's 16x128 x-tile (f32->f16) into [32K,64K),
  //      build persistent A fragments (ISA 16-bit A 16x32 layout) ----
  Frag afrag[4];
  {
    unsigned* dstA = (unsigned*)(ldsMem + 32768) + wave * 1024;
    const float4* src = (const float4*)(x + (size_t)rowBase * DIM);
#pragma unroll
    for (int i = 0; i < 16; ++i) {
      int q = lane + 32 * i;                 // float4 index in contiguous tile
      float4 v = src[q];
      dstA[2 * q]     = pack_f16(v.x, v.y);
      dstA[2 * q + 1] = pack_f16(v.z, v.w);
    }
    // lane l, frag f, vgpr j: K = 32f + 16*(j>>2) + 2*(j&3) + 8h
    const unsigned* a = dstA + m * 64;
#pragma unroll
    for (int f = 0; f < 4; ++f) {
      afrag[f].q[0] = *(const uint4*)(a + 16 * f + 4 * h);
      afrag[f].q[1] = *(const uint4*)(a + 16 * f + 8 + 4 * h);
    }
  }

  float best[8];
  int   bidx[8];
#pragma unroll
  for (int r = 0; r < 8; ++r) { best[r] = -INFINITY; bidx[r] = 0; }

  wait_asynccnt0();     // buf0 landed
  __syncthreads();      // everyone consumed the A region; buf0 visible to all

  for (int ci = 0; ci < NCHUNKS; ++ci) {
    const int cbase = colStart + ci * NCHUNK;
    const unsigned curOff = (ci & 1) ? 32768u : 0u;

    // ---- prefetch next chunk into the other buffer (hidden behind WMMAs) ----
    if (ci + 1 < NCHUNKS) {
      const unsigned nxtOff = curOff ^ 32768u;
      const char* g = (const char*)(cbh + (size_t)(cbase + NCHUNK) * DIM);
#pragma unroll
      for (int r = 0; r < 8; ++r)
        async_copy_b128(ldsBase + nxtOff + tid * 16 + r * 4096,
                        g + tid * 16 + r * 4096);
    }

    const unsigned char* bufp = ldsMem + curOff;
#pragma unroll
    for (int t = 0; t < 8; ++t) {
      // B fragments (ISA 16-bit B 32x16 layout): lane = column n = 16t + m,
      // vgpr j: K = 32f + 16*(j>>2) + 2*(j&3) + 8h => uint index = 16f + 8*(j>>2) + (j&3) + 4h
      const unsigned* b = (const unsigned*)(bufp + (size_t)(16 * t + m) * 256);
      Frag bfrag[4];
#pragma unroll
      for (int f = 0; f < 4; ++f) {
        bfrag[f].q[0] = *(const uint4*)(b + 16 * f + 4 * h);
        bfrag[f].q[1] = *(const uint4*)(b + 16 * f + 8 + 4 * h);
      }

      v8f c = {};
#pragma unroll
      for (int f = 0; f < 4; ++f)
        c = __builtin_amdgcn_wmma_f32_16x16x32_f16(
            false, afrag[f].v, false, bfrag[f].v, (short)0, c, false, false);

      // C/D layout: vgpr r of lane l holds score[row = r + 8h][col = 16t + m]
      const int col = cbase + 16 * t + m;
#pragma unroll
      for (int r = 0; r < 8; ++r) {
        if (c[r] > best[r]) { best[r] = c[r]; bidx[r] = col; }
      }
    }

    wait_asynccnt0();   // my async writes for the next buffer are in LDS
    __syncthreads();    // all waves done reading curOff / writing nxtOff
  }

  // ---- reduce (max, first-index) across the 16 lanes sharing each row ----
#pragma unroll
  for (int r = 0; r < 8; ++r) {
    float v = best[r];
    int   i = bidx[r];
#pragma unroll
    for (int mask = 8; mask >= 1; mask >>= 1) {
      float ov = __shfl_xor(v, mask, 32);
      int   oi = __shfl_xor(i, mask, 32);
      if (ov > v || (ov == v && oi < i)) { v = ov; i = oi; }
    }
    if (m == 0) {
      int row = rowBase + r + 8 * h;
      unsigned eb = __float_as_uint(v);
      eb ^= (eb & 0x80000000u) ? 0xFFFFFFFFu : 0x80000000u;
      unsigned long long packed =
          ((unsigned long long)eb << 32) |
          (unsigned long long)(0xFFFFFFFFu - (unsigned)i);
      atomicMax(ws + row, packed);
    }
  }
}

// =====================================================================
// Fallback (scratch too small for f16 codebook): round-1 kernel with
// in-loop f32->f16 conversion. Same math, same output.
// =====================================================================
__global__ __launch_bounds__(THREADS)
void lfq_wmma_argmax_cvt(const float* __restrict__ x,
                         const float* __restrict__ cb,
                         unsigned long long* __restrict__ ws) {
  __shared__ unsigned ldsA[8 * 1024];
  __shared__ unsigned ldsB[8192];

  const int tid  = threadIdx.x;
  const int wave = tid >> 5;
  const int lane = tid & 31;
  const int h    = lane >> 4;
  const int m    = lane & 15;

  const int rowBase  = blockIdx.x * MTILE + wave * 16;
  const int colStart = blockIdx.y * COLSPAN;
  const int colEnd   = colStart + COLSPAN;

  {
    const float4* src = (const float4*)(x + (size_t)rowBase * DIM);
    unsigned* dst = ldsA + wave * 1024;
#pragma unroll
    for (int i = 0; i < 16; ++i) {
      int q = lane + 32 * i;
      float4 v = src[q];
      dst[2 * q]     = pack_f16(v.x, v.y);
      dst[2 * q + 1] = pack_f16(v.z, v.w);
    }
  }

  Frag afrag[4];
  {
    const unsigned* a = ldsA + wave * 1024 + m * 64;
#pragma unroll
    for (int f = 0; f < 4; ++f) {
      afrag[f].q[0] = *(const uint4*)(a + 16 * f + 4 * h);
      afrag[f].q[1] = *(const uint4*)(a + 16 * f + 8 + 4 * h);
    }
  }

  float best[8];
  int   bidx[8];
#pragma unroll
  for (int r = 0; r < 8; ++r) { best[r] = -INFINITY; bidx[r] = 0; }

  for (int cbase = colStart; cbase < colEnd; cbase += NCHUNK) {
    __syncthreads();
    {
      const float4* src = (const float4*)(cb + (size_t)cbase * DIM);
#pragma unroll
      for (int i = 0; i < 16; ++i) {
        int q = tid + THREADS * i;
        float4 v = src[q];
        ldsB[2 * q]     = pack_f16(v.x, v.y);
        ldsB[2 * q + 1] = pack_f16(v.z, v.w);
      }
    }
    __syncthreads();

#pragma unroll
    for (int t = 0; t < 8; ++t) {
      const unsigned* b = ldsB + (16 * t + m) * 64;
      Frag bfrag[4];
#pragma unroll
      for (int f = 0; f < 4; ++f) {
        bfrag[f].q[0] = *(const uint4*)(b + 16 * f + 4 * h);
        bfrag[f].q[1] = *(const uint4*)(b + 16 * f + 8 + 4 * h);
      }
      v8f c = {};
#pragma unroll
      for (int f = 0; f < 4; ++f)
        c = __builtin_amdgcn_wmma_f32_16x16x32_f16(
            false, afrag[f].v, false, bfrag[f].v, (short)0, c, false, false);
      const int col = cbase + 16 * t + m;
#pragma unroll
      for (int r = 0; r < 8; ++r) {
        if (c[r] > best[r]) { best[r] = c[r]; bidx[r] = col; }
      }
    }
  }

#pragma unroll
  for (int r = 0; r < 8; ++r) {
    float v = best[r];
    int   i = bidx[r];
#pragma unroll
    for (int mask = 8; mask >= 1; mask >>= 1) {
      float ov = __shfl_xor(v, mask, 32);
      int   oi = __shfl_xor(i, mask, 32);
      if (ov > v || (ov == v && oi < i)) { v = ov; i = oi; }
    }
    if (m == 0) {
      int row = rowBase + r + 8 * h;
      unsigned eb = __float_as_uint(v);
      eb ^= (eb & 0x80000000u) ? 0xFFFFFFFFu : 0x80000000u;
      unsigned long long packed =
          ((unsigned long long)eb << 32) |
          (unsigned long long)(0xFFFFFFFFu - (unsigned)i);
      atomicMax(ws + row, packed);
    }
  }
}

// ---- small helper kernels ----
__global__ void lfq_init_ws(unsigned long long* ws) {
  int i = blockIdx.x * blockDim.x + threadIdx.x;
  if (i < NTOKENS) ws[i] = 0ull;
}

__global__ void lfq_cvt_cb(const float* __restrict__ cb, __half* __restrict__ cbh) {
  int i = blockIdx.x * blockDim.x + threadIdx.x;   // float4 index
  float4 v = ((const float4*)cb)[i];
  ((__half2*)cbh)[2 * i]     = __floats2half2_rn(v.x, v.y);
  ((__half2*)cbh)[2 * i + 1] = __floats2half2_rn(v.z, v.w);
}

__global__ void lfq_finalize(const unsigned long long* __restrict__ ws,
                             int* __restrict__ out_idx,
                             float* __restrict__ out_loss) {
  int i = blockIdx.x * blockDim.x + threadIdx.x;
  if (i < NTOKENS)
    out_idx[i] = (int)(0xFFFFFFFFu - (unsigned)(ws[i] & 0xFFFFFFFFull));
  if (i == 0) out_loss[0] = 0.0f;  // reference returns constant 0.0 loss
}

extern "C" void kernel_launch(void* const* d_in, const int* in_sizes, int n_in,
                              void* d_out, int out_size, void* d_ws, size_t ws_size,
                              hipStream_t stream) {
  const float* x  = (const float*)d_in[0];   // [16384, 128] f32
  const float* cb = (const float*)d_in[1];   // [16384, 128] f32

  unsigned long long* ws = (unsigned long long*)d_ws;       // 128 KB packed (score,~idx)
  const size_t packBytes = (size_t)NTOKENS * sizeof(unsigned long long);
  __half* cbh = (__half*)((char*)d_ws + packBytes);         // 4 MB f16 codebook
  const size_t need = packBytes + (size_t)NCODES * DIM * sizeof(__half);

  int*   out_idx  = (int*)d_out;
  float* out_loss = (float*)d_out + NTOKENS;

  lfq_init_ws<<<NTOKENS / 256, 256, 0, stream>>>(ws);

  dim3 grid(NTOKENS / MTILE, NSPLIT);
  if (ws_size >= need) {
    lfq_cvt_cb<<<(NCODES * DIM / 4) / 256, 256, 0, stream>>>(cb, cbh);
    lfq_wmma_argmax_async<<<grid, THREADS, 0, stream>>>(x, cbh, ws);
  } else {
    lfq_wmma_argmax_cvt<<<grid, THREADS, 0, stream>>>(x, cb, ws);
  }

  lfq_finalize<<<NTOKENS / 256, 256, 0, stream>>>(ws, out_idx, out_loss);
}
